// SSD_74096775791049
// MI455X (gfx1250) — matrix operand order repaired and checked by reading, not compiled, not run on hardware
//
#include <hip/hip_runtime.h>
#include <hip/hip_bf16.h>

#define D_MODEL 1024
#define HEADDIM 128
#define D_STATE 64
#define NHEADS  8
#define NPROJ   (NHEADS + 2*D_STATE)   // 136
#define T_LEN   2048
#define BATCH   2
#define CHUNK   64
#define PW      32
#define PT      (HEADDIM/PW)           // 4
#define LDPAD   80                     // bf16 row stride: 160 B, 16B-aligned rows

typedef __bf16 bf16;
typedef __attribute__((ext_vector_type(16))) __bf16 v16bf;
typedef __attribute__((ext_vector_type(8)))  float  v8f;

union FragU { v16bf v; float4 f[2]; };

// ---------------- WMMA fragment helpers (CDNA5 ISA 7.12.2 layouts, wave32) ----
// All operands laid out so each lane reads two contiguous 16-byte runs
// -> compiler emits ds_load_b128 pairs.

// A matrix 16x32 bf16, element (m,k) at base[m*stride + k]; base & stride*2 16B-aligned
__device__ __forceinline__ v16bf load_a_frag(const bf16* base, int stride) {
  int lane = threadIdx.x & 31;
  int m = lane & 15, half = lane >> 4;
  const bf16* rowp = base + m * stride;
  FragU u;
  u.f[0] = *(const float4*)(rowp + half * 8);        // VGPR0..3: k = half*8 .. +7
  u.f[1] = *(const float4*)(rowp + 16 + half * 8);   // VGPR4..7: k = 16+half*8 .. +7
  return u.v;
}

// B matrix 32x16 bf16 from transposed storage: element (k,n) at base[n*stride + k]
__device__ __forceinline__ v16bf load_b_frag_t(const bf16* base, int stride) {
  int lane = threadIdx.x & 31;
  int n = lane & 15, half = lane >> 4;
  const bf16* rowp = base + n * stride;
  FragU u;
  u.f[0] = *(const float4*)(rowp + half * 8);        // k = half*8 .. +7
  u.f[1] = *(const float4*)(rowp + 16 + half * 8);   // k = 16+half*8 .. +7
  return u.v;
}

__device__ __forceinline__ v8f wmma_bf16f32(v16bf a, v16bf b, v8f c) {
  return __builtin_amdgcn_wmma_f32_16x16x32_bf16(false, a, false, b, (short)0, c,
                                                 false, false);
}

// ---------------- Kernel 1: tiled fp32 projection GEMM + gating -------------
// 32 token rows x 136 outputs per block, K staged in LDS (W reused 32x).
#define ROWS 32
#define KT   64

__global__ void __launch_bounds__(256) ssd_proj_kernel(
    const float* __restrict__ x, const float* __restrict__ W,
    const float* __restrict__ dt_bias, const float* __restrict__ A_log,
    bf16* __restrict__ Xh, bf16* __restrict__ Bg, bf16* __restrict__ Cg,
    float* __restrict__ Araw)
{
  __shared__ float xs[ROWS][KT + 1];        // 8.3 KB
  __shared__ float ws[NPROJ][KT + 1];       // 35.4 KB
  __shared__ float proj[ROWS][NPROJ + 1];   // 17.5 KB
  __shared__ float dts[ROWS][NHEADS];       // 1 KB

  int tid  = threadIdx.x;
  int row0 = blockIdx.x * ROWS;             // global token index base

  float acc[17];                            // 32*136 = 4352 = 17*256
  int rI[17], cI[17];
#pragma unroll
  for (int u = 0; u < 17; ++u) {
    acc[u] = 0.f;
    int idx = tid + 256 * u;
    rI[u] = idx / NPROJ;
    cI[u] = idx % NPROJ;
  }

  for (int kt = 0; kt < D_MODEL; kt += KT) {
    for (int i = tid; i < ROWS * KT; i += 256) {
      int r = i >> 6, k = i & 63;
      xs[r][k] = x[(size_t)(row0 + r) * D_MODEL + kt + k];
    }
    for (int i = tid; i < NPROJ * KT; i += 256) {
      int c = i >> 6, k = i & 63;
      ws[c][k] = W[(size_t)c * D_MODEL + kt + k];
      if (kt + KT < D_MODEL)
        __builtin_prefetch(&W[(size_t)c * D_MODEL + kt + KT + k], 0, 1);
    }
    __syncthreads();
#pragma unroll
    for (int u = 0; u < 17; ++u) {
      const float* xr = xs[rI[u]];
      const float* wr = ws[cI[u]];
      float a = acc[u];
#pragma unroll 8
      for (int k = 0; k < KT; ++k) a = fmaf(xr[k], wr[k], a);
      acc[u] = a;
    }
    __syncthreads();
  }

#pragma unroll
  for (int u = 0; u < 17; ++u) proj[rI[u]][cI[u]] = acc[u];
  __syncthreads();

  {
    int r = tid >> 3, h = tid & 7;          // 256 = 32 rows * 8 heads
    float z  = proj[r][h] + dt_bias[h];
    float dt = (z > 20.f) ? z : log1pf(__expf(z));   // softplus
    dts[r][h] = dt;
    int grow = row0 + r;
    Araw[((size_t)(grow / T_LEN) * NHEADS + h) * T_LEN + (grow % T_LEN)] =
        -__expf(A_log[h]) * dt;
  }
  __syncthreads();

  for (int i = tid; i < ROWS * D_STATE; i += 256) {
    int r = i >> 6, n = i & 63;
    Bg[(size_t)(row0 + r) * D_STATE + n] = (bf16)proj[r][NHEADS + n];
    Cg[(size_t)(row0 + r) * D_STATE + n] = (bf16)proj[r][NHEADS + D_STATE + n];
  }
  for (int i = tid; i < ROWS * D_MODEL; i += 256) {
    int r = i >> 10, col = i & 1023, h = col >> 7;
    Xh[(size_t)(row0 + r) * D_MODEL + col] =
        (bf16)(x[(size_t)(row0 + r) * D_MODEL + col] * dts[r][h]);
  }
}

// ---------------- Kernel 2: per-(b,h) inclusive cumsum of a -----------------
__global__ void ssd_cumsum_kernel(const float* __restrict__ Araw,
                                  float* __restrict__ Acs)
{
  if (threadIdx.x != 0) return;
  size_t base = (size_t)blockIdx.x * T_LEN;
  float c = 0.f;
  for (int t = 0; t < T_LEN; ++t) { c += Araw[base + t]; Acs[base + t] = c; }
}

// ---------------- Kernel 3: chunked SSD core (WMMA) -------------------------
__global__ void __launch_bounds__(256) ssd_core_kernel(
    const bf16* __restrict__ Xh, const bf16* __restrict__ Bg,
    const bf16* __restrict__ Cg, const float* __restrict__ Acs,
    const float* __restrict__ x, const float* __restrict__ Dp,
    float* __restrict__ out)
{
  __shared__ __align__(16) bf16  sC[CHUNK][LDPAD];      // C chunk (later C*e^r)
  __shared__ __align__(16) bf16  sB[CHUNK][LDPAD];      // B chunk [l][n]
  __shared__ __align__(16) bf16  sG[CHUNK][LDPAD];      // decayed scores [l][s]
  __shared__ __align__(16) bf16  sBt[D_STATE][LDPAD];   // (B*e^{rT-r})^T [n][l]
  __shared__ __align__(16) bf16  sXt[PW][LDPAD];        // X chunk transposed [p][l]
  __shared__ __align__(16) bf16  sSbt[PW][LDPAD];       // state bf16 transposed [p][n]
  __shared__ float sS[D_STATE][PW + 1];                 // state f32 [n][p]
  __shared__ float sr[CHUNK];

  int tid  = threadIdx.x;
  int wave = tid >> 5;
  int lane = tid & 31;
  int m16  = lane & 15, half = lane >> 4;

  int blk = blockIdx.x;
  int b   = blk / (NHEADS * PT);
  int hp  = blk % (NHEADS * PT);
  int h   = hp / PT;
  int pt  = hp % PT;
  float Dh = Dp[h];

  for (int i = tid; i < D_STATE * PW; i += 256) {
    int n = i >> 5, p = i & 31;
    sS[n][p]   = 0.f;
    sSbt[p][n] = (bf16)0.f;
  }
  __syncthreads();

  size_t caBase = ((size_t)b * NHEADS + h) * T_LEN;

  for (int q = 0; q < T_LEN / CHUNK; ++q) {
    int t0 = q * CHUNK;

    // ---- stage 1: load chunk inputs ----
    float rbase = (q == 0) ? 0.f : Acs[caBase + t0 - 1];
    for (int i = tid; i < CHUNK * D_STATE; i += 256) {
      int l = i >> 6, n = i & 63;
      size_t g = ((size_t)b * T_LEN + t0 + l) * D_STATE + n;
      sC[l][n] = Cg[g];
      sB[l][n] = Bg[g];
    }
    for (int i = tid; i < CHUNK * PW; i += 256) {
      int l = i >> 5, p = i & 31;
      sXt[p][l] = Xh[((size_t)b * T_LEN + t0 + l) * D_MODEL
                     + h * HEADDIM + pt * PW + p];
    }
    if (tid < CHUNK) sr[tid] = Acs[caBase + t0 + tid] - rbase;
    __syncthreads();
    float rT = sr[CHUNK - 1];

    // ---- stage 2: G = tril( (C B^T) * exp(r_l - r_s) ), 16 tiles / 8 waves --
#pragma unroll
    for (int rep = 0; rep < 2; ++rep) {
      int tile = wave + rep * 8;             // 0..15
      int ti = tile >> 2, tj = tile & 3;     // l-tile, s-tile
      if (tj > ti) {
#pragma unroll
        for (int rr = 0; rr < 8; ++rr)
          sG[ti * 16 + half * 8 + rr][tj * 16 + m16] = (bf16)0.f;
      } else {
        v8f acc = {};
        acc = wmma_bf16f32(load_a_frag(&sC[ti * 16][0], LDPAD),
                           load_b_frag_t(&sB[tj * 16][0], LDPAD), acc);
        acc = wmma_bf16f32(load_a_frag(&sC[ti * 16][32], LDPAD),
                           load_b_frag_t(&sB[tj * 16][32], LDPAD), acc);
#pragma unroll
        for (int rr = 0; rr < 8; ++rr) {
          int l = ti * 16 + half * 8 + rr;
          int s = tj * 16 + m16;
          float v = (s <= l) ? acc[rr] * __expf(sr[l] - sr[s]) : 0.f;
          sG[l][s] = (bf16)v;
        }
      }
    }
    __syncthreads();

    // ---- stage 3: scale C in place (C*e^r), build Bt = (B*e^{rT-r})^T ----
    for (int i = tid; i < CHUNK * D_STATE; i += 256) {
      int l = i >> 6, n = i & 63;
      sC[l][n]  = (bf16)((float)sC[l][n] * __expf(sr[l]));
      sBt[n][l] = (bf16)((float)sB[l][n] * __expf(rT - sr[l]));
    }
    __syncthreads();

    // ---- stage 4: Y = G @ X + (C e^r) @ S ; write out with D skip ----
    {
      int ti = wave >> 1, tp = wave & 1;     // l-tile (0..3), p-tile (0..1)
      v8f acc = {};
#pragma unroll
      for (int kk = 0; kk < CHUNK; kk += 32)
        acc = wmma_bf16f32(load_a_frag(&sG[ti * 16][kk], LDPAD),
                           load_b_frag_t(&sXt[tp * 16][kk], LDPAD), acc);
#pragma unroll
      for (int kk = 0; kk < D_STATE; kk += 32)
        acc = wmma_bf16f32(load_a_frag(&sC[ti * 16][kk], LDPAD),
                           load_b_frag_t(&sSbt[tp * 16][kk], LDPAD), acc);
#pragma unroll
      for (int rr = 0; rr < 8; ++rr) {
        int l = ti * 16 + half * 8 + rr;
        int p = tp * 16 + m16;
        size_t gi = ((size_t)b * T_LEN + t0 + l) * D_MODEL
                    + h * HEADDIM + pt * PW + p;
        out[gi] = acc[rr] + x[gi] * Dh;
      }
    }
    __syncthreads();

    // ---- stage 5: S = e^{rT} * S + Bt @ X ----
    {
      int ni = wave >> 1, tp = wave & 1;     // n-tile (0..3), p-tile (0..1)
      v8f acc = {};
#pragma unroll
      for (int kk = 0; kk < CHUNK; kk += 32)
        acc = wmma_bf16f32(load_a_frag(&sBt[ni * 16][kk], LDPAD),
                           load_b_frag_t(&sXt[tp * 16][kk], LDPAD), acc);
      float erT = __expf(rT);
#pragma unroll
      for (int rr = 0; rr < 8; ++rr) {
        int n = ni * 16 + half * 8 + rr;
        int p = tp * 16 + m16;
        float ns = erT * sS[n][p] + acc[rr];
        sS[n][p]   = ns;
        sSbt[p][n] = (bf16)ns;
      }
    }
    __syncthreads();
  }
}

// ---------------- host launch ----------------
extern "C" void kernel_launch(void* const* d_in, const int* in_sizes, int n_in,
                              void* d_out, int out_size, void* d_ws, size_t ws_size,
                              hipStream_t stream) {
  (void)in_sizes; (void)n_in; (void)out_size; (void)ws_size;
  const float* x       = (const float*)d_in[0];
  const float* W       = (const float*)d_in[1];
  const float* dt_bias = (const float*)d_in[2];
  const float* A_log   = (const float*)d_in[3];
  const float* Dp      = (const float*)d_in[4];
  float* out = (float*)d_out;

  const size_t nX  = (size_t)BATCH * T_LEN * D_MODEL;   // bf16
  const size_t nBC = (size_t)BATCH * T_LEN * D_STATE;   // bf16 each
  const size_t nA  = (size_t)BATCH * NHEADS * T_LEN;    // f32 each
  auto align256 = [](size_t v) { return (v + 255) & ~(size_t)255; };

  uint8_t* p = (uint8_t*)d_ws;
  bf16*  Xh   = (bf16*)p;  p += align256(nX  * sizeof(bf16));
  bf16*  Bg   = (bf16*)p;  p += align256(nBC * sizeof(bf16));
  bf16*  Cg   = (bf16*)p;  p += align256(nBC * sizeof(bf16));
  float* Araw = (float*)p; p += align256(nA  * sizeof(float));
  float* Acs  = (float*)p;

  ssd_proj_kernel<<<(BATCH * T_LEN) / ROWS, 256, 0, stream>>>(
      x, W, dt_bias, A_log, Xh, Bg, Cg, Araw);
  ssd_cumsum_kernel<<<BATCH * NHEADS, 32, 0, stream>>>(Araw, Acs);
  ssd_core_kernel<<<BATCH * NHEADS * PT, 256, 0, stream>>>(
      Xh, Bg, Cg, Acs, x, Dp, out);
}
